// HybridHyperedgeGenerator_17549236371596
// MI455X (gfx1250) — matrix operand order repaired and checked by reading, not compiled, not run on hardware
//
#include <hip/hip_runtime.h>
#include <hip/hip_bf16.h>

typedef __bf16 bf16;
typedef __attribute__((ext_vector_type(16))) bf16  v16bf;
typedef __attribute__((ext_vector_type(8)))  bf16  v8bf;
typedef __attribute__((ext_vector_type(8)))  float v8f;
typedef __attribute__((ext_vector_type(4)))  int   v4i;

#define NN   8192
#define D0_  512
#define D1_  768
#define HID_ 512
#define KTOP 16
#define KP1  17

#if __has_builtin(__builtin_amdgcn_global_load_async_to_lds_b128)
#define HAVE_ASYNC_LDS 1
#endif

// 16-byte global -> LDS copy; async (ASYNCcnt) when available, else via VGPRs.
__device__ __forceinline__ void copy16_to_lds(const void* g, void* l) {
#ifdef HAVE_ASYNC_LDS
  __builtin_amdgcn_global_load_async_to_lds_b128(
      (__attribute__((address_space(1))) v4i*)(g),
      (__attribute__((address_space(3))) v4i*)(l),
      /*imm offset*/ 0, /*cpol*/ 0);
#else
  *(v8bf*)l = *(const v8bf*)g;
#endif
}
__device__ __forceinline__ void copy_to_lds_join() {
#ifdef HAVE_ASYNC_LDS
#if __has_builtin(__builtin_amdgcn_s_wait_asynccnt)
  __builtin_amdgcn_s_wait_asynccnt(0);
#else
  asm volatile("s_wait_asynccnt 0x0" ::: "memory");
#endif
#endif
}

// ---------------------------------------------------------------------------
// elementwise fp32 -> bf16
__global__ void f32_to_bf16_kernel(const float* __restrict__ src,
                                   bf16* __restrict__ dst, size_t n) {
  size_t i = (size_t)blockIdx.x * blockDim.x + threadIdx.x;
  size_t stride = (size_t)gridDim.x * blockDim.x;
  for (; i < n; i += stride) dst[i] = (bf16)src[i];
}

// transpose + convert: src[rows][cols] f32 -> dst[cols][rows] bf16
__global__ void transpose_to_bf16_kernel(const float* __restrict__ src,
                                         bf16* __restrict__ dst,
                                         int rows, int cols) {
  size_t n = (size_t)rows * cols;
  size_t i = (size_t)blockIdx.x * blockDim.x + threadIdx.x;
  size_t stride = (size_t)gridDim.x * blockDim.x;
  for (; i < n; i += stride) {
    int r = (int)(i / cols), c = (int)(i % cols);
    dst[(size_t)c * rows + r] = (bf16)src[i];
  }
}

// ---------------------------------------------------------------------------
// Generic bf16 WMMA GEMM:  out = act(A[M,Kd] x Bt[Nc,Kd]^T + bias)
// Block: 8 waves = 128 rows x 16 cols. The 16-col B tile is shared by all 8
// waves, so it is staged once into LDS (async copy), cutting its global
// traffic 8x; A streams from global (rows are wave-private).
__global__ __launch_bounds__(256)
void gemm_bf16_kernel(const bf16* __restrict__ A, const bf16* __restrict__ Bt,
                      const float* __restrict__ bias,
                      float* __restrict__ outF, bf16* __restrict__ outB,
                      int M, int Kd, int Nc, int relu) {
  __shared__ __align__(16) bf16 ldsB[16 * 768];  // max Kd = 768 -> 24 KB
  const int lane = threadIdx.x & 31;
  const int wave = threadIdx.x >> 5;
  const int n0 = blockIdx.x * 16;
  const int m0 = (blockIdx.y * 8 + wave) * 16;  // grids are exact multiples

  // stage B tile (rows n0..n0+15 of Bt are one contiguous 16*Kd block)
  {
    const char* src = (const char*)(Bt + (size_t)n0 * Kd);
    char* dst = (char*)ldsB;
    const int nchunks = (16 * Kd * (int)sizeof(bf16)) / 16;
    for (int c16 = threadIdx.x; c16 < nchunks; c16 += 256)
      copy16_to_lds(src + (size_t)c16 * 16, dst + (size_t)c16 * 16);
    copy_to_lds_join();
  }
  __syncthreads();

  const int row  = m0 + (lane & 15);
  const int aOff = (lane < 16) ? 0 : 16;  // A K-segment base (contiguous k = 16h + s)
  const int bOff = (lane < 16) ? 0 : 16;  // B K-segment base (contiguous k = 16h + s)

  const bf16* ap  = A + (size_t)row * Kd + aOff;
  const bf16* bpl = ldsB + (size_t)(lane & 15) * Kd + bOff;

  v8f acc = {};
#pragma unroll 4
  for (int kb = 0; kb < Kd; kb += 32) {
    v8bf alo = *(const v8bf*)(ap + kb);
    v8bf ahi = *(const v8bf*)(ap + kb + 8);
    v16bf a = __builtin_shufflevector(alo, ahi, 0,1,2,3,4,5,6,7,
                                      8,9,10,11,12,13,14,15);
    v16bf b = *(const v16bf*)(bpl + kb);
    acc = __builtin_amdgcn_wmma_f32_16x16x32_bf16(
        false, a, false, b, (short)0, acc, false, false);
  }

  const int rb = m0 + ((lane < 16) ? 0 : 8);
  const int c  = n0 + (lane & 15);
  const float bv = bias ? bias[c] : 0.0f;
#pragma unroll
  for (int i = 0; i < 8; ++i) {
    float v = acc[i] + bv;
    if (relu) v = fmaxf(v, 0.0f);
    const size_t o = (size_t)(rb + i) * Nc + c;
    if (outF) outF[o] = v;
    if (outB) outB[o] = (bf16)v;
  }
}

// ---------------------------------------------------------------------------
// fused = (a0*f0 + a1*f1) as bf16, with softmax(attn_w) computed on device
__global__ void fuse_kernel(const float* __restrict__ f0,
                            const float* __restrict__ f1,
                            const float* __restrict__ attn,
                            bf16* __restrict__ pre, size_t n) {
  float w0 = attn[0], w1 = attn[1];
  float m = fmaxf(w0, w1);
  float e0 = __expf(w0 - m), e1 = __expf(w1 - m);
  float a0 = e0 / (e0 + e1), a1 = e1 / (e0 + e1);
  size_t i = (size_t)blockIdx.x * blockDim.x + threadIdx.x;
  size_t stride = (size_t)gridDim.x * blockDim.x;
  for (; i < n; i += stride) pre[i] = (bf16)(a0 * f0[i] + a1 * f1[i]);
}

// per-row L2 normalize (to bf16) + row mean
__global__ __launch_bounds__(256)
void norm_mean_kernel(const float* __restrict__ fused,
                      bf16* __restrict__ normed,
                      float* __restrict__ row_mean, int Nc) {
  __shared__ float ssq[256], ssm[256];
  const int r = blockIdx.x;
  const int tid = threadIdx.x;
  float sq = 0.f, sm = 0.f;
  for (int c = tid; c < Nc; c += 256) {
    float v = fused[(size_t)r * Nc + c];
    sq += v * v; sm += v;
  }
  ssq[tid] = sq; ssm[tid] = sm;
  __syncthreads();
  for (int s = 128; s > 0; s >>= 1) {
    if (tid < s) { ssq[tid] += ssq[tid + s]; ssm[tid] += ssm[tid + s]; }
    __syncthreads();
  }
  float inv = 1.0f / fmaxf(sqrtf(ssq[0]), 1e-12f);
  if (tid == 0) row_mean[r] = ssm[0] / (float)Nc;
  for (int c = tid; c < Nc; c += 256)
    normed[(size_t)r * Nc + c] = (bf16)(fused[(size_t)r * Nc + c] * inv);
}

// ---------------------------------------------------------------------------
// sim = normed @ normed^T fused with streaming top-17 per row.
// Block: 256 threads (8 waves) owns 16 rows; iterates 128-column slabs.
// The 16-row A block is shared by all 8 waves -> staged once in LDS (async).
// normed row-major [N,HID] is exactly the Bt layout needed for WMMA B.
__global__ __launch_bounds__(256)
void sim_topk_kernel(const bf16* __restrict__ normed,
                     int* __restrict__ edges, int N, int Kd) {
  __shared__ __align__(16) bf16 ldsA[16 * 512];  // 16 KB (Kd == 512)
  __shared__ float tile[16][132];  // pad 128->132: avoid LDS bank conflicts
  const int lane = threadIdx.x & 31;
  const int wave = threadIdx.x >> 5;
  const int m0 = blockIdx.x * 16;

  // stage the 16-row A block (contiguous 16*Kd bf16) into LDS
  {
    const char* src = (const char*)(normed + (size_t)m0 * Kd);
    char* dst = (char*)ldsA;
    const int nchunks = (16 * Kd * (int)sizeof(bf16)) / 16;
    for (int c16 = threadIdx.x; c16 < nchunks; c16 += 256)
      copy16_to_lds(src + (size_t)c16 * 16, dst + (size_t)c16 * 16);
    copy_to_lds_join();
  }
  __syncthreads();

  const int aOff = (lane < 16) ? 0 : 16;
  const int bOff = (lane < 16) ? 0 : 16;
  const bf16* apl = ldsA + (size_t)(lane & 15) * Kd + aOff;

  float tv[KP1]; int ti[KP1];
#pragma unroll
  for (int j = 0; j < KP1; ++j) { tv[j] = -1e30f; ti[j] = -1; }

  for (int cbase = 0; cbase < N; cbase += 128) {
    const int col = cbase + wave * 16 + (lane & 15);
    const bf16* bp = normed + (size_t)col * Kd + bOff;
    if (cbase + 128 < N)  // pull next slab's B rows toward L2/L0
      __builtin_prefetch(bp + (size_t)128 * Kd, 0, 1);
    v8f acc = {};
#pragma unroll 4
    for (int kb = 0; kb < Kd; kb += 32) {
      v8bf alo = *(const v8bf*)(apl + kb);
      v8bf ahi = *(const v8bf*)(apl + kb + 8);
      v16bf a = __builtin_shufflevector(alo, ahi, 0,1,2,3,4,5,6,7,
                                        8,9,10,11,12,13,14,15);
      v16bf b = *(const v16bf*)(bp + kb);
      acc = __builtin_amdgcn_wmma_f32_16x16x32_bf16(
          false, a, false, b, (short)0, acc, false, false);
    }
    __syncthreads();  // previous slab's top-k scan finished
    const int c  = wave * 16 + (lane & 15);
    const int rb = (lane < 16) ? 0 : 8;
#pragma unroll
    for (int i = 0; i < 8; ++i) tile[rb + i][c] = acc[i];
    __syncthreads();
    if (threadIdx.x < 16) {
      for (int c2 = 0; c2 < 128; ++c2) {
        float v = tile[threadIdx.x][c2];
        if (v > tv[KP1 - 1]) {              // strict '>' keeps stable tie order
          int j = KP1 - 1;
          while (j > 0 && v > tv[j - 1]) {
            tv[j] = tv[j - 1]; ti[j] = ti[j - 1]; --j;
          }
          tv[j] = v; ti[j] = cbase + c2;
        }
      }
    }
  }
  if (threadIdx.x < 16) {
    const int grow = m0 + threadIdx.x;
    int outv[KP1]; outv[0] = grow;
    int cnt = 1;
    for (int j = 0; j < KP1 && cnt < KP1; ++j)
      if (ti[j] != grow) outv[cnt++] = ti[j];
    for (int t = 0; t < KP1; ++t) edges[(size_t)grow * KP1 + t] = outv[t];
  }
}

// ---------------------------------------------------------------------------
__global__ void edge_weight_kernel(const int* __restrict__ edges,
                                   const float* __restrict__ row_mean,
                                   float* __restrict__ weights, int N) {
  int r = blockIdx.x * blockDim.x + threadIdx.x;
  if (r >= N) return;
  float s = 0.f;
#pragma unroll
  for (int j = 0; j < KP1; ++j) s += row_mean[edges[(size_t)r * KP1 + j]];
  float w = 1.0f / (1.0f + __expf(-(s / (float)KP1)));
  weights[r] = (w > 0.5f) ? w : 0.0f;
}

__global__ void zero_f32_kernel(float4* __restrict__ p, size_t n4) {
  size_t i = (size_t)blockIdx.x * blockDim.x + threadIdx.x;
  size_t stride = (size_t)gridDim.x * blockDim.x;
  float4 z = make_float4(0.f, 0.f, 0.f, 0.f);
  for (; i < n4; i += stride) p[i] = z;
}

__global__ void scatter_kernel(const int* __restrict__ edges,
                               const float* __restrict__ weights,
                               float* __restrict__ H, int N) {
  int t = blockIdx.x * blockDim.x + threadIdx.x;
  if (t >= N * KP1) return;
  int r = t / KP1;
  if (weights[r] > 0.0f) H[(size_t)edges[t] * N + r] = 1.0f;
}

// ---------------------------------------------------------------------------
static inline size_t align256(size_t x) { return (x + 255) & ~(size_t)255; }

extern "C" void kernel_launch(void* const* d_in, const int* in_sizes, int n_in,
                              void* d_out, int out_size, void* d_ws, size_t ws_size,
                              hipStream_t stream) {
  (void)in_sizes; (void)n_in; (void)out_size; (void)ws_size;
  const float* x0   = (const float*)d_in[0];
  const float* x1   = (const float*)d_in[1];
  const float* w1_0 = (const float*)d_in[2];
  const float* b1_0 = (const float*)d_in[3];
  const float* w2_0 = (const float*)d_in[4];
  const float* b2_0 = (const float*)d_in[5];
  const float* w1_1 = (const float*)d_in[6];
  const float* b1_1 = (const float*)d_in[7];
  const float* w2_1 = (const float*)d_in[8];
  const float* b2_1 = (const float*)d_in[9];
  const float* attn = (const float*)d_in[10];
  const float* fw   = (const float*)d_in[11];
  const float* fb   = (const float*)d_in[12];

  const int N = NN, D0 = D0_, D1 = D1_, HID = HID_;

  char* ws = (char*)d_ws;
  size_t off = 0;
  auto carve = [&](size_t bytes) -> void* {
    void* p = ws + off; off = align256(off + bytes); return p;
  };
  bf16*  x0b   = (bf16*)carve((size_t)N * D0 * sizeof(bf16));
  bf16*  x1b   = (bf16*)carve((size_t)N * D1 * sizeof(bf16));
  bf16*  w10t  = (bf16*)carve((size_t)HID * D0 * sizeof(bf16));
  bf16*  w20t  = (bf16*)carve((size_t)HID * HID * sizeof(bf16));
  bf16*  w11t  = (bf16*)carve((size_t)HID * D1 * sizeof(bf16));
  bf16*  w21t  = (bf16*)carve((size_t)HID * HID * sizeof(bf16));
  bf16*  fwt   = (bf16*)carve((size_t)HID * HID * sizeof(bf16));
  bf16*  h0b   = (bf16*)carve((size_t)N * HID * sizeof(bf16));
  float* f0    = (float*)carve((size_t)N * HID * sizeof(float));
  float* f1    = (float*)carve((size_t)N * HID * sizeof(float));
  float* rmean = (float*)carve((size_t)N * sizeof(float));
  int*   edges = (int*)  carve((size_t)N * KP1 * sizeof(int));
  // aliases (lifetimes are disjoint; same-stream kernels serialize)
  bf16*  preb    = x0b;   // x0b dead after first GEMM
  bf16*  h1b     = h0b;   // h0b dead after f0 GEMM
  bf16*  normedb = x1b;   // x1b dead after modality-1 GEMM
  float* fusedf  = f0;    // f0 dead after fuse

  float* H       = (float*)d_out;
  float* weights = H + (size_t)N * N;

  // 1) convert inputs / pre-transpose weights to bf16
  f32_to_bf16_kernel<<<2048, 256, 0, stream>>>(x0, x0b, (size_t)N * D0);
  f32_to_bf16_kernel<<<2048, 256, 0, stream>>>(x1, x1b, (size_t)N * D1);
  transpose_to_bf16_kernel<<<512, 256, 0, stream>>>(w1_0, w10t, D0, HID);
  transpose_to_bf16_kernel<<<512, 256, 0, stream>>>(w2_0, w20t, HID, HID);
  transpose_to_bf16_kernel<<<768, 256, 0, stream>>>(w1_1, w11t, D1, HID);
  transpose_to_bf16_kernel<<<512, 256, 0, stream>>>(w2_1, w21t, HID, HID);
  transpose_to_bf16_kernel<<<512, 256, 0, stream>>>(fw,   fwt,  HID, HID);

  // 2) modality MLPs (WMMA bf16, f32 accumulate)
  dim3 blk(256);
  dim3 g512(HID / 16, N / 128);
  gemm_bf16_kernel<<<g512, blk, 0, stream>>>(x0b, w10t, b1_0, nullptr, h0b,
                                             N, D0, HID, 1);
  gemm_bf16_kernel<<<g512, blk, 0, stream>>>(h0b, w20t, b2_0, f0, nullptr,
                                             N, HID, HID, 0);
  gemm_bf16_kernel<<<g512, blk, 0, stream>>>(x1b, w11t, b1_1, nullptr, h1b,
                                             N, D1, HID, 1);
  gemm_bf16_kernel<<<g512, blk, 0, stream>>>(h1b, w21t, b2_1, f1, nullptr,
                                             N, HID, HID, 0);

  // 3) attention fuse + fusion GEMM
  fuse_kernel<<<2048, 256, 0, stream>>>(f0, f1, attn, preb, (size_t)N * HID);
  gemm_bf16_kernel<<<g512, blk, 0, stream>>>(preb, fwt, fb, fusedf, nullptr,
                                             N, HID, HID, 0);

  // 4) normalize rows + row means
  norm_mean_kernel<<<N, 256, 0, stream>>>(fusedf, normedb, rmean, HID);

  // 5) cosine-sim GEMM fused with streaming top-17 (never materialize sim)
  sim_topk_kernel<<<N / 16, 256, 0, stream>>>(normedb, edges, N, HID);

  // 6) edge weights, zero H, scatter incidence
  edge_weight_kernel<<<(N + 255) / 256, 256, 0, stream>>>(edges, rmean,
                                                          weights, N);
  zero_f32_kernel<<<8192, 256, 0, stream>>>((float4*)H, (size_t)N * N / 4);
  scatter_kernel<<<(N * KP1 + 255) / 256, 256, 0, stream>>>(edges, weights,
                                                            H, N);
}
